// SpMiddleFHD_AUX_82729660055577
// MI455X (gfx1250) — matrix operand order repaired and checked by reading, not compiled, not run on hardware
//
#include <hip/hip_runtime.h>
#include <hip/hip_bf16.h>
#include <cstdint>

// ---------------------------------------------------------------------------
// CDNA5 (gfx1250) wave32 WMMA types
// ---------------------------------------------------------------------------
typedef __attribute__((ext_vector_type(16))) __bf16       v16bf;
typedef __attribute__((ext_vector_type(8)))  float        v8f;
typedef __attribute__((ext_vector_type(8)))  unsigned int v8u;
typedef __attribute__((ext_vector_type(4)))  unsigned int v4u;

__device__ __forceinline__ unsigned short f32_to_bf16_bits(float f) {
  union { float f; unsigned u; } x; x.f = f;
  unsigned r = x.u + 0x7fffu + ((x.u >> 16) & 1u);   // round-nearest-even
  return (unsigned short)(r >> 16);
}

// ---------------------------------------------------------------------------
// Utility fills
// ---------------------------------------------------------------------------
__global__ void fill_f32(float* p, size_t n, float v) {
  size_t i = (size_t)blockIdx.x * blockDim.x + threadIdx.x;
  if (i < n) p[i] = v;
}
__global__ void fill_i32(int* p, size_t n, int v) {
  size_t i = (size_t)blockIdx.x * blockDim.x + threadIdx.x;
  if (i < n) p[i] = v;
}

// voxel -> row-index grid (spconv rulebook equivalent)
__global__ void scatter_grid(const int* __restrict__ coords, int N,
                             int* __restrict__ grid, int D, int H, int W) {
  int i = blockIdx.x * blockDim.x + threadIdx.x;
  if (i >= N) return;
  int b = coords[i * 4 + 0], z = coords[i * 4 + 1];
  int y = coords[i * 4 + 2], x = coords[i * 4 + 3];
  grid[(((size_t)b * D + z) * H + y) * W + x] = i;
}

// f32 features -> bf16, channel-padded with zeros
__global__ void feats_to_bf16(const float* __restrict__ x,
                              unsigned short* __restrict__ xb,
                              long total, int C, int Cpad) {
  long idx = (long)blockIdx.x * blockDim.x + threadIdx.x;
  if (idx >= total) return;
  int  c = (int)(idx % Cpad);
  long i = idx / Cpad;
  float v = (c < C) ? x[i * C + c] : 0.f;
  xb[idx] = f32_to_bf16_bits(v);
}

// f32 weight [nk][Cin][Cout] -> bf16 [nk][Cout][Cin_pad]  (K-contiguous B layout)
__global__ void prep_weight(const float* __restrict__ w,
                            unsigned short* __restrict__ wt,
                            int nk, int Cin, int Cout, int Cin_pad) {
  long idx = (long)blockIdx.x * blockDim.x + threadIdx.x;
  long total = (long)nk * Cout * Cin_pad;
  if (idx >= total) return;
  int ci = (int)(idx % Cin_pad);
  long t = idx / Cin_pad;
  int co = (int)(t % Cout);
  int k  = (int)(t / Cout);
  float v = (ci < Cin) ? w[((size_t)k * Cin + ci) * Cout + co] : 0.f;
  wt[idx] = f32_to_bf16_bits(v);
}

// ---------------------------------------------------------------------------
// Fused gather + WMMA GEMM sparse conv.
// Block = 128 threads (4 waves). Each wave owns 16 output rows x up to 64
// output columns (4 accumulator tiles) -> A gathered ONCE per K-chunk and
// reused across 4 B tiles. Per kernel-offset, the block cooperatively stages
// the 64-col x Cin_pad bf16 weight slice into LDS (<= 8 KB), and B fragments
// are served from LDS (ds_load_b128).
//   nk==27, stride2==0 : SubMConv3d k=3      (outputs at input sites)
//   nk==27, stride2==1 : SparseConv3d k=3 s=2 p=1 (in = out*2-1+kofs)
//   nk==1              : 1x1x1 conv (identity gather via grid)
// WMMA layouts (wave32):
//   A (16x32 bf16): lane L = row L&15; lanes<16 K={0..7,16..23},
//                   lanes>=16 K={8..15,24..31}
//   B (32x16 bf16): lane L = col L&15; lanes<16 K={0..7,16..23},
//                   lanes>=16 K={8..15,24..31}
//   C/D (16x16 f32): VGPR i -> M = i + 8*hi, N = lane&15
// EXEC stays all-ones (no early returns; uniform k loop / barriers).
// ---------------------------------------------------------------------------
__global__ __launch_bounds__(128)
void spconv_wmma_bf16(const unsigned short* __restrict__ featsb, // Nin x Cin_pad
                      const int* __restrict__ ocoords,           // Nout x 4
                      const int* __restrict__ vgrid,             // B,D,H,W
                      int D, int H, int W,
                      const unsigned short* __restrict__ wt,     // nk x Cout x Cin_pad
                      float* __restrict__ out,                   // Nout x Cout
                      int Nout, int Cin_pad, int Cout, int nk, int stride2) {
  __shared__ __align__(32) unsigned short smem[64 * 64];   // 8 KB weight stage

  const int tid  = threadIdx.x;
  const int lane = tid & 31;
  const int wv   = tid >> 5;            // wave id 0..3
  const int mrow = lane & 15;
  const int hi   = lane >> 4;
  const int row0 = (blockIdx.x << 6) + (wv << 4);
  const int col0 = blockIdx.y << 6;     // 64-column group
  int nct = (Cout - col0) >> 4;         // number of 16-col tiles this group
  if (nct > 4) nct = 4;

  v8f acc[4];
#pragma unroll
  for (int ct = 0; ct < 4; ++ct)
    acc[ct] = (v8f){0.f, 0.f, 0.f, 0.f, 0.f, 0.f, 0.f, 0.f};

  int r = row0 + mrow;
  int cb = -1, cz = 0, cy = 0, cx = 0;
  if (r < Nout) {
    cb = ocoords[r * 4 + 0];
    cz = ocoords[r * 4 + 1];
    cy = ocoords[r * 4 + 2];
    cx = ocoords[r * 4 + 3];
  }

  const int ncols   = nct << 4;                 // columns staged (16/32/48/64)
  const int stage_n = (ncols * Cin_pad) >> 3;   // v4u (16B) elements to stage

  for (int k = 0; k < nk; ++k) {
    // ---- cooperative LDS stage of this offset's weight slice --------------
    const unsigned short* wslice =
        wt + ((size_t)k * Cout + col0) * Cin_pad;
    const v4u* gsrc = (const v4u*)(const void*)wslice;
    v4u* ldst = (v4u*)(void*)smem;
    for (int i = tid; i < stage_n; i += 128) ldst[i] = gsrc[i];
    __syncthreads();

    // ---- neighbor index for this wave's 16 rows ---------------------------
    int zz, yy, xx;
    if (nk == 1) { zz = cz; yy = cy; xx = cx; }
    else {
      int dz = k / 9, dy = (k / 3) % 3, dx = k % 3;        // 0..2
      if (stride2) { zz = cz * 2 - 1 + dz; yy = cy * 2 - 1 + dy; xx = cx * 2 - 1 + dx; }
      else         { zz = cz + dz - 1;     yy = cy + dy - 1;     xx = cx + dx - 1; }
    }
    int nidx = -1;
    if (cb >= 0 && zz >= 0 && zz < D && yy >= 0 && yy < H && xx >= 0 && xx < W)
      nidx = vgrid[(((size_t)cb * D + zz) * H + yy) * W + xx];

    const unsigned short* frow =
        featsb + (nidx >= 0 ? (size_t)nidx : 0) * Cin_pad + hi * 8;
    if (k + 1 < nk)                                        // global_prefetch_b8
      __builtin_prefetch(wt + ((size_t)(k + 1) * Cout + col0) * Cin_pad, 0, 1);

    for (int kc = 0; kc < Cin_pad; kc += 32) {
      // A fragment: gathered once, reused by all 4 column tiles
      v16bf afrag;
      if (nidx >= 0) {
        v4u lo = *(const v4u*)(const void*)(frow + kc);        // K run 1 (8 bf16)
        v4u hh = *(const v4u*)(const void*)(frow + kc + 16);   // K run 2 (8 bf16)
        v8u au;
#pragma unroll
        for (int i = 0; i < 4; ++i) { au[i] = lo[i]; au[i + 4] = hh[i]; }
        afrag = __builtin_bit_cast(v16bf, au);
      } else {
        v8u z = {0u, 0u, 0u, 0u, 0u, 0u, 0u, 0u};
        afrag = __builtin_bit_cast(v16bf, z);
      }
#pragma unroll
      for (int ct = 0; ct < 4; ++ct) {
        if (ct < nct) {
          const unsigned short* bp =
              &smem[((ct << 4) + mrow) * Cin_pad + kc + hi * 8];
          v4u blo = *(const v4u*)(const void*)bp;
          v4u bhh = *(const v4u*)(const void*)(bp + 16);
          v8u bu;
#pragma unroll
          for (int i = 0; i < 4; ++i) { bu[i] = blo[i]; bu[i + 4] = bhh[i]; }
          v16bf bfrag = __builtin_bit_cast(v16bf, bu);
          acc[ct] = __builtin_amdgcn_wmma_f32_16x16x32_bf16(
              /*neg_a=*/false, afrag, /*neg_b=*/false, bfrag,
              /*c_mod=*/(short)0, acc[ct], /*reuse_a=*/false, /*reuse_b=*/false);
        }
      }
    }
    __syncthreads();   // protect smem before next offset's stage
  }

#pragma unroll
  for (int ct = 0; ct < 4; ++ct) {
    if (ct < nct) {
#pragma unroll
      for (int i = 0; i < 8; ++i) {      // D layout: VGPR i -> M=i+hi*8, N=mrow
        int rr = row0 + i + hi * 8;
        if (rr < Nout)
          out[(size_t)rr * Cout + col0 + (ct << 4) + mrow] = acc[ct][i];
      }
    }
  }
}

// ---------------------------------------------------------------------------
// Training-mode BatchNorm1d (eps=1e-3) + ReLU
// ---------------------------------------------------------------------------
__global__ void bn_stats(const float* __restrict__ x, int N, int C,
                         float* __restrict__ mu, float* __restrict__ rs) {
  int c = blockIdx.x * blockDim.x + threadIdx.x;
  if (c >= C) return;
  float s = 0.f, s2 = 0.f;
  for (int i = 0; i < N; ++i) { float v = x[(size_t)i * C + c]; s += v; s2 += v * v; }
  float m = s / (float)N;
  float var = s2 / (float)N - m * m;
  mu[c] = m;
  rs[c] = rsqrtf(var + 1e-3f);
}

__global__ void bn_relu_apply(const float* __restrict__ x, float* __restrict__ y,
                              unsigned short* __restrict__ yb,
                              const float* __restrict__ mu, const float* __restrict__ rs,
                              const float* __restrict__ ga, const float* __restrict__ be,
                              int N, int C, int Cpad) {
  long idx = (long)blockIdx.x * blockDim.x + threadIdx.x;
  if (idx >= (long)N * Cpad) return;
  int  c = (int)(idx % Cpad);
  long i = idx / Cpad;
  float v = 0.f;
  if (c < C) {
    v = (x[(size_t)i * C + c] - mu[c]) * rs[c] * ga[c] + be[c];
    v = v > 0.f ? v : 0.f;
    y[(size_t)i * C + c] = v;
  }
  if (yb) yb[idx] = f32_to_bf16_bits(v);
}

// ---------------------------------------------------------------------------
// tensor2points / points_mean / three-NN interpolation
// ---------------------------------------------------------------------------
__global__ void points_mean_k(const int* __restrict__ coords,
                              const float* __restrict__ vf, int N,
                              float* __restrict__ pm) {
  int i = blockIdx.x * blockDim.x + threadIdx.x;
  if (i >= N) return;
  pm[i * 4 + 0] = (float)coords[i * 4 + 0];
  pm[i * 4 + 1] = vf[i * 4 + 0];
  pm[i * 4 + 2] = vf[i * 4 + 1];
  pm[i * 4 + 3] = vf[i * 4 + 2];
}

__global__ void t2p(const int* __restrict__ coords, int N,
                    float vx, float vy, float vz, float* __restrict__ out) {
  int i = blockIdx.x * blockDim.x + threadIdx.x;
  if (i >= N) return;
  out[i * 4 + 0] = (float)coords[i * 4 + 0];
  out[i * 4 + 1] = (float)coords[i * 4 + 3] * vx + 0.0f  + 0.5f * vx;
  out[i * 4 + 2] = (float)coords[i * 4 + 2] * vy - 40.0f + 0.5f * vy;
  out[i * 4 + 3] = (float)coords[i * 4 + 1] * vz - 3.0f  + 0.5f * vz;
}

__global__ void nn_interp(const float* __restrict__ unk, int Nu,
                          const float* __restrict__ kno, int Nk,
                          const float* __restrict__ kf, int C,
                          float* __restrict__ out, int ostride, int ocol0) {
  int i = blockIdx.x * blockDim.x + threadIdx.x;
  if (i >= Nu) return;
  float ub = unk[i * 4 + 0], ux = unk[i * 4 + 1];
  float uy = unk[i * 4 + 2], uz = unk[i * 4 + 3];
  float d0 = 1e30f, d1 = 1e30f, dv2 = 1e30f;
  int   i0 = 0,     i1 = 0,     i2 = 0;
  for (int j = 0; j < Nk; ++j) {
    float dx = ux - kno[j * 4 + 1];
    float dy = uy - kno[j * 4 + 2];
    float dz = uz - kno[j * 4 + 3];
    float d = dx * dx + dy * dy + dz * dz;
    if (ub != kno[j * 4 + 0]) d += 1e10f;             // batch mismatch penalty
    if (d < d0)       { dv2 = d1; i2 = i1; d1 = d0; i1 = i0; d0 = d; i0 = j; }
    else if (d < d1)  { dv2 = d1; i2 = i1; d1 = d;  i1 = j; }
    else if (d < dv2) { dv2 = d;  i2 = j; }
  }
  float w0 = 1.f / (d0 + 1e-8f), w1 = 1.f / (d1 + 1e-8f), w2 = 1.f / (dv2 + 1e-8f);
  float ws = w0 + w1 + w2;
  w0 /= ws; w1 /= ws; w2 /= ws;
  for (int c = 0; c < C; ++c)
    out[(size_t)i * ostride + ocol0 + c] =
        w0 * kf[(size_t)i0 * C + c] + w1 * kf[(size_t)i1 * C + c] +
        w2 * kf[(size_t)i2 * C + c];
}

// dense (B, C*D, H, W) scatter of level-3 features (C=320, D=6, H=50, W=44)
__global__ void scatter_dense(const float* __restrict__ x,
                              const int* __restrict__ coords, int N,
                              float* __restrict__ dense) {
  long idx = (long)blockIdx.x * blockDim.x + threadIdx.x;
  if (idx >= (long)N * 320) return;
  int  c = (int)(idx % 320);
  long i = idx / 320;
  int b = coords[i * 4 + 0], z = coords[i * 4 + 1];
  int y = coords[i * 4 + 2], xw = coords[i * 4 + 3];
  dense[(((size_t)b * 1920 + (size_t)c * 6 + z) * 50 + y) * 44 + xw] = x[idx];
}

// small dense GEMM for the point head
__global__ void dense_mm(const float* __restrict__ A, const float* __restrict__ Bm,
                         float* __restrict__ C, int M, int K, int N) {
  long idx = (long)blockIdx.x * blockDim.x + threadIdx.x;
  if (idx >= (long)M * N) return;
  int  j = (int)(idx % N);
  long i = idx / N;
  float s = 0.f;
  for (int k = 0; k < K; ++k) s += A[i * K + k] * Bm[(size_t)k * N + j];
  C[idx] = s;
}

// ---------------------------------------------------------------------------
// Host wiring
// ---------------------------------------------------------------------------
static inline size_t alignup256(size_t x) { return (x + 255u) & ~(size_t)255u; }
static inline unsigned nblk(long n, int b) { return (unsigned)((n + b - 1) / b); }

extern "C" void kernel_launch(void* const* d_in, const int* in_sizes, int n_in,
                              void* d_out, int out_size, void* d_ws, size_t ws_size,
                              hipStream_t stream) {
  (void)n_in; (void)ws_size;
  const float* vf = (const float*)d_in[0];
  const int* c0 = (const int*)d_in[1];
  const int* c1 = (const int*)d_in[2];
  const int* c2 = (const int*)d_in[3];
  const int* c3 = (const int*)d_in[4];
  const int N0 = in_sizes[0] / 4, N1 = in_sizes[2] / 4;
  const int N2 = in_sizes[3] / 4, N3 = in_sizes[4] / 4;
  const int B = 2;
  const long DENSE_SZ = (long)B * 1920 * 50 * 44;   // (2, 320*6, 50, 44)

  // ---- workspace carve-up --------------------------------------------------
  char* wsp = (char*)d_ws; size_t off = 0;
  auto carve = [&](size_t bytes) -> void* {
    void* p = wsp + off; off = alignup256(off + bytes); return p;
  };
  int* g0 = (int*)carve((size_t)B * 41 * 400 * 352 * 4);
  int* g1 = (int*)carve((size_t)B * 21 * 200 * 176 * 4);
  int* g2 = (int*)carve((size_t)B * 11 * 100 * 88 * 4);
  int* g3 = (int*)carve((size_t)B * 6 * 50 * 44 * 4);
  size_t featCap = (size_t)N0 * 64;
  if ((size_t)N3 * 320 > featCap) featCap = (size_t)N3 * 320;
  float*          convout = (float*)carve(featCap * 4);
  float*          xpost   = (float*)carve(featCap * 4);
  unsigned short* xb      = (unsigned short*)carve((size_t)N0 * 64 * 2);
  unsigned short* wt      = (unsigned short*)carve((size_t)27 * 64 * 64 * 2);
  float*          bn_mu   = (float*)carve(320 * 4);
  float*          bn_rs   = (float*)carve(320 * 4);
  float*          pfeat   = (float*)carve((size_t)N0 * 160 * 4);
  float*          pw      = (float*)carve((size_t)N0 * 64 * 4);
  int nkmax = N1 > N2 ? N1 : N2; if (N3 > nkmax) nkmax = N3;
  float*          kno     = (float*)carve((size_t)nkmax * 4 * 4);

  float* pm_out  = (float*)d_out + DENSE_SZ;                 // points_mean (N0 x 4)
  float* cls_out = pm_out + (size_t)N0 * 4;                  // point_cls   (N0 x 1)
  float* reg_out = cls_out + (size_t)N0;                     // point_reg   (N0 x 3)

  // ---- zero output, build grids -------------------------------------------
  fill_f32<<<nblk((long)out_size, 256), 256, 0, stream>>>((float*)d_out, (size_t)out_size, 0.f);
  size_t ng0 = (size_t)B * 41 * 400 * 352, ng1 = (size_t)B * 21 * 200 * 176;
  size_t ng2 = (size_t)B * 11 * 100 * 88,  ng3 = (size_t)B * 6 * 50 * 44;
  fill_i32<<<nblk((long)ng0, 256), 256, 0, stream>>>(g0, ng0, -1);
  fill_i32<<<nblk((long)ng1, 256), 256, 0, stream>>>(g1, ng1, -1);
  fill_i32<<<nblk((long)ng2, 256), 256, 0, stream>>>(g2, ng2, -1);
  fill_i32<<<nblk((long)ng3, 256), 256, 0, stream>>>(g3, ng3, -1);
  scatter_grid<<<nblk(N0, 256), 256, 0, stream>>>(c0, N0, g0, 41, 400, 352);
  scatter_grid<<<nblk(N1, 256), 256, 0, stream>>>(c1, N1, g1, 21, 200, 176);
  scatter_grid<<<nblk(N2, 256), 256, 0, stream>>>(c2, N2, g2, 11, 100, 88);
  scatter_grid<<<nblk(N3, 256), 256, 0, stream>>>(c3, N3, g3, 6, 50, 44);

  points_mean_k<<<nblk(N0, 256), 256, 0, stream>>>(c0, vf, N0, pm_out);

  // ---- conv + bn helpers ---------------------------------------------------
  auto run_conv = [&](const unsigned short* fin, const int* oc, const int* vg,
                      int D, int H, int W, const float* w,
                      int Cin, int Cout, int nk, int s2, int Nout) {
    int Cin_pad = (Cin <= 32) ? 32 : 64;
    prep_weight<<<nblk((long)nk * Cout * Cin_pad, 256), 256, 0, stream>>>(
        w, wt, nk, Cin, Cout, Cin_pad);
    dim3 g((Nout + 63) / 64, (Cout + 63) / 64);
    spconv_wmma_bf16<<<g, 128, 0, stream>>>(fin, oc, vg, D, H, W, wt, convout,
                                            Nout, Cin_pad, Cout, nk, s2);
  };
  auto run_bn = [&](int N, int C, const float* ga, const float* be, int write_bf) {
    bn_stats<<<nblk(C, 64), 64, 0, stream>>>(convout, N, C, bn_mu, bn_rs);
    int Cp = write_bf ? ((C <= 32) ? 32 : 64) : C;
    bn_relu_apply<<<nblk((long)N * Cp, 256), 256, 0, stream>>>(
        convout, xpost, write_bf ? xb : (unsigned short*)nullptr,
        bn_mu, bn_rs, ga, be, N, C, Cp);
  };
  auto P = [&](int i) { return (const float*)d_in[i]; };

  // ---- network -------------------------------------------------------------
  feats_to_bf16<<<nblk((long)N0 * 32, 256), 256, 0, stream>>>(vf, xb, (long)N0 * 32, 4, 32);

  // conv0: subm 4->16, 16->16
  run_conv(xb, c0, g0, 41, 400, 352, P(6), 4, 16, 27, 0, N0);  run_bn(N0, 16, P(7),  P(8),  1);
  run_conv(xb, c0, g0, 41, 400, 352, P(9), 16, 16, 27, 0, N0); run_bn(N0, 16, P(10), P(11), 1);
  // down0: stride 16->32 (level0 -> level1)
  run_conv(xb, c1, g0, 41, 400, 352, P(12), 16, 32, 27, 1, N1); run_bn(N1, 32, P(13), P(14), 1);
  // conv1: subm 32->32 x2
  run_conv(xb, c1, g1, 21, 200, 176, P(15), 32, 32, 27, 0, N1); run_bn(N1, 32, P(16), P(17), 1);
  run_conv(xb, c1, g1, 21, 200, 176, P(18), 32, 32, 27, 0, N1); run_bn(N1, 32, P(19), P(20), 1);
  // p1
  t2p<<<nblk(N1, 256), 256, 0, stream>>>(c1, N1, 0.1f, 0.1f, 0.2f, kno);
  nn_interp<<<nblk(N0, 128), 128, 0, stream>>>(pm_out, N0, kno, N1, xpost, 32, pfeat, 160, 0);
  // down1: stride 32->64
  run_conv(xb, c2, g1, 21, 200, 176, P(21), 32, 64, 27, 1, N2); run_bn(N2, 64, P(22), P(23), 1);
  // conv2: subm 64->64 x3
  run_conv(xb, c2, g2, 11, 100, 88, P(24), 64, 64, 27, 0, N2);  run_bn(N2, 64, P(25), P(26), 1);
  run_conv(xb, c2, g2, 11, 100, 88, P(27), 64, 64, 27, 0, N2);  run_bn(N2, 64, P(28), P(29), 1);
  run_conv(xb, c2, g2, 11, 100, 88, P(30), 64, 64, 27, 0, N2);  run_bn(N2, 64, P(31), P(32), 1);
  // p2
  t2p<<<nblk(N2, 256), 256, 0, stream>>>(c2, N2, 0.2f, 0.2f, 0.4f, kno);
  nn_interp<<<nblk(N0, 128), 128, 0, stream>>>(pm_out, N0, kno, N2, xpost, 64, pfeat, 160, 32);
  // down2: stride 64->64
  run_conv(xb, c3, g2, 11, 100, 88, P(33), 64, 64, 27, 1, N3);  run_bn(N3, 64, P(34), P(35), 1);
  // conv3: subm 64->64 x3
  run_conv(xb, c3, g3, 6, 50, 44, P(36), 64, 64, 27, 0, N3);    run_bn(N3, 64, P(37), P(38), 1);
  run_conv(xb, c3, g3, 6, 50, 44, P(39), 64, 64, 27, 0, N3);    run_bn(N3, 64, P(40), P(41), 1);
  run_conv(xb, c3, g3, 6, 50, 44, P(42), 64, 64, 27, 0, N3);    run_bn(N3, 64, P(43), P(44), 1);
  // p3
  t2p<<<nblk(N3, 256), 256, 0, stream>>>(c3, N3, 0.4f, 0.4f, 0.8f, kno);
  nn_interp<<<nblk(N0, 128), 128, 0, stream>>>(pm_out, N0, kno, N3, xpost, 64, pfeat, 160, 96);
  // extra: 1x1x1 64->320 (identity gather via grid, nk=1)
  run_conv(xb, c3, g3, 6, 50, 44, P(45), 64, 320, 1, 0, N3);    run_bn(N3, 320, P(46), P(47), 0);
  // dense scatter
  scatter_dense<<<nblk((long)N3 * 320, 256), 256, 0, stream>>>(xpost, c3, N3, (float*)d_out);
  // point head
  dense_mm<<<nblk((long)N0 * 64, 256), 256, 0, stream>>>(pfeat, P(48), pw, N0, 160, 64);
  dense_mm<<<nblk((long)N0 * 1, 256), 256, 0, stream>>>(pw, P(49), cls_out, N0, 64, 1);
  dense_mm<<<nblk((long)N0 * 3, 256), 256, 0, stream>>>(pw, P(50), reg_out, N0, 64, 3);
}
